// CGConvNet_87290915324233
// MI455X (gfx1250) — matrix operand order repaired and checked by reading, not compiled, not run on hardware
//
#include <hip/hip_runtime.h>

typedef __attribute__((ext_vector_type(16))) _Float16 v16h;
typedef __attribute__((ext_vector_type(8)))  float    v8f;

#define NN 50000
#define NE 800000
#define EP (NE + NN)   // edges + self loops

// ---------------- math helpers ----------------
__device__ inline float sigmoidf_(float x) { return 1.0f / (1.0f + __expf(-x)); }
__device__ inline float softplusf_(float x) { return x > 20.0f ? x : log1pf(__expf(x)); }
__device__ inline float lrelu02_(float x) { return x > 0.0f ? x : 0.2f * x; }
__device__ inline unsigned encf_(float f) {
    unsigned u = __float_as_uint(f);
    return (u & 0x80000000u) ? ~u : (u | 0x80000000u);
}
__device__ inline float decf_(unsigned e) {
    return __uint_as_float((e & 0x80000000u) ? (e ^ 0x80000000u) : ~e);
}

// ---------------- WMMA fragment loaders ----------------
// A fragment (16x32 f16, M=rows, K from LDS row-major with fixed stride 96):
// lane (m = lane&15, h = lane>>4) holds K = k0+16h .. k0+16h+15 of row m.
__device__ inline v16h load_a_frag(const _Float16* z, int k0, int lane) {
    const int m = lane & 15, h = lane >> 4;
    const _Float16* p0 = z + m * 96 + k0 + 16 * h;
    v16h a;
#pragma unroll
    for (int i = 0; i < 16; i++) a[i] = p0[i];
    return a;
}
// B fragment (32x16 f16, K x N) from column-major weights (stride Kpad, multiple of 32):
// lane (n = n0 + lane&15, h = lane>>4) holds K = k0+16h .. k0+16h+15 of column n.
__device__ inline v16h load_b_frag(const _Float16* w, int Kpad, int n0, int k0, int lane) {
    const int n = n0 + (lane & 15), h = lane >> 4;
    const _Float16* p = w + n * Kpad + k0 + 16 * h;
    v16h b;
#pragma unroll
    for (int i = 0; i < 16; i++) b[i] = p[i];
    return b;
}

// ---------------- weight convert: row-major f32 [Kin,NC] -> col-major f16 [NC][Kpad] ----------------
__global__ void k_convw(const float* __restrict__ W, _Float16* __restrict__ w16,
                        int Kin, int Kpad, int NC) {
    int i = blockIdx.x * blockDim.x + threadIdx.x;
    if (i >= NC * Kpad) return;
    int n = i / Kpad, k = i % Kpad;
    w16[i] = (k < Kin) ? (_Float16)W[k * NC + n] : (_Float16)0.0f;
}

// ---------------- fused CGConv edge kernel ----------------
// Per wave: 16 edges. z = [x[dst], x[src], e] (72 -> pad 96) staged in LDS as f16.
// msg = sigmoid(z@Wf+bf) * softplus(z@Ws+bs); atomicAdd into agg[dst].
__global__ __launch_bounds__(128) void k_cgconv_edges(
    const float* __restrict__ x, const float* __restrict__ ea,
    const int* __restrict__ src, const int* __restrict__ dst,
    const _Float16* __restrict__ wfh, const _Float16* __restrict__ wsh,
    const float* __restrict__ bf, const float* __restrict__ bs,
    float* __restrict__ agg)
{
    __shared__ _Float16 zt[4][16 * 96];
    const int wave = threadIdx.x >> 5, lane = threadIdx.x & 31;
    const int e0 = (blockIdx.x * 4 + wave) * 16;
    const int r = lane & 15, p = lane >> 4;
    const int e = e0 + r;
    _Float16* zr = &zt[wave][r * 96];
    if (p == 0) {
        if (e < NE) {
            const float4* xp = (const float4*)(x + (size_t)dst[e] * 32);
#pragma unroll
            for (int i = 0; i < 8; i++) {
                float4 v = xp[i];
                zr[i*4+0]=(_Float16)v.x; zr[i*4+1]=(_Float16)v.y;
                zr[i*4+2]=(_Float16)v.z; zr[i*4+3]=(_Float16)v.w;
            }
        } else {
#pragma unroll
            for (int i = 0; i < 32; i++) zr[i] = (_Float16)0.0f;
        }
    } else {
        if (e < NE) {
            const float4* xp = (const float4*)(x + (size_t)src[e] * 32);
#pragma unroll
            for (int i = 0; i < 8; i++) {
                float4 v = xp[i];
                zr[32+i*4+0]=(_Float16)v.x; zr[32+i*4+1]=(_Float16)v.y;
                zr[32+i*4+2]=(_Float16)v.z; zr[32+i*4+3]=(_Float16)v.w;
            }
            const float4* epn = (const float4*)(ea + (size_t)e * 8);
#pragma unroll
            for (int i = 0; i < 2; i++) {
                float4 v = epn[i];
                zr[64+i*4+0]=(_Float16)v.x; zr[64+i*4+1]=(_Float16)v.y;
                zr[64+i*4+2]=(_Float16)v.z; zr[64+i*4+3]=(_Float16)v.w;
            }
        } else {
            for (int i = 32; i < 72; i++) zr[i] = (_Float16)0.0f;
        }
#pragma unroll
        for (int i = 72; i < 96; i++) zr[i] = (_Float16)0.0f;
    }
    __syncthreads();

    v8f aF0 = {}, aF1 = {}, aS0 = {}, aS1 = {};
#pragma unroll
    for (int kc = 0; kc < 3; kc++) {
        const int k0 = kc * 32;
        v16h a = load_a_frag(zt[wave], k0, lane);
        v16h b;
        b = load_b_frag(wfh, 96, 0, k0, lane);
        aF0 = __builtin_amdgcn_wmma_f32_16x16x32_f16(false, a, false, b, (short)0, aF0, false, false);
        b = load_b_frag(wfh, 96, 16, k0, lane);
        aF1 = __builtin_amdgcn_wmma_f32_16x16x32_f16(false, a, false, b, (short)0, aF1, false, false);
        b = load_b_frag(wsh, 96, 0, k0, lane);
        aS0 = __builtin_amdgcn_wmma_f32_16x16x32_f16(false, a, false, b, (short)0, aS0, false, false);
        b = load_b_frag(wsh, 96, 16, k0, lane);
        aS1 = __builtin_amdgcn_wmma_f32_16x16x32_f16(false, a, false, b, (short)0, aS1, false, false);
    }

    const int n = lane & 15, hh = lane >> 4;
    const float bf0 = bf[n], bf1 = bf[16 + n], bs0 = bs[n], bs1 = bs[16 + n];
#pragma unroll
    for (int v = 0; v < 8; v++) {
        const int ee = e0 + v + 8 * hh;
        if (ee < NE) {
            const int d = dst[ee];
            float m0 = sigmoidf_(aF0[v] + bf0) * softplusf_(aS0[v] + bs0);
            float m1 = sigmoidf_(aF1[v] + bf1) * softplusf_(aS1[v] + bs1);
            atomicAdd(&agg[(size_t)d * 32 + n],      m0);
            atomicAdd(&agg[(size_t)d * 32 + 16 + n], m1);
        }
    }
}

// out = relu(a + b), elementwise
__global__ void k_add_relu(const float* __restrict__ a, const float* __restrict__ b,
                           float* __restrict__ out, int n) {
    int i = blockIdx.x * blockDim.x + threadIdx.x;
    if (i < n) { float v = a[i] + b[i]; out[i] = v > 0.0f ? v : 0.0f; }
}

// ---------------- generic WMMA GEMM: Hout[M,NC] = X[M,K] @ W16colmajor[K,NC] ----------------
// K in {32,96} (multiple of 32, <=96), NC multiple of 16.
__global__ __launch_bounds__(128) void k_gemm_wmma(
    const float* __restrict__ X, const _Float16* __restrict__ W,
    float* __restrict__ Hout, int M, int K, int NC)
{
    __shared__ _Float16 at[4][16 * 96];
    const int wave = threadIdx.x >> 5, lane = threadIdx.x & 31;
    const int row0 = (blockIdx.x * 4 + wave) * 16;
    const int r = lane & 15, p = lane >> 4;
    const int row = row0 + r;
    _Float16* ar = &at[wave][r * 96];
    const int half = K >> 1;
    if (row < M) {
        const float* xp = X + (size_t)row * K + p * half;
        for (int i = 0; i < half; i += 4) {
            float4 v = *(const float4*)(xp + i);
            ar[p*half+i+0]=(_Float16)v.x; ar[p*half+i+1]=(_Float16)v.y;
            ar[p*half+i+2]=(_Float16)v.z; ar[p*half+i+3]=(_Float16)v.w;
        }
    } else {
        for (int i = 0; i < half; i++) ar[p * half + i] = (_Float16)0.0f;
    }
    __syncthreads();

    const int n_lo = lane & 15, hh = lane >> 4;
    for (int t = 0; t < NC; t += 16) {
        v8f acc = {};
        for (int k0 = 0; k0 < K; k0 += 32) {
            v16h a = load_a_frag(at[wave], k0, lane);
            v16h b = load_b_frag(W, K, t, k0, lane);
            acc = __builtin_amdgcn_wmma_f32_16x16x32_f16(false, a, false, b, (short)0, acc, false, false);
        }
#pragma unroll
        for (int v = 0; v < 8; v++) {
            const int rr = row0 + v + 8 * hh;
            if (rr < M) Hout[(size_t)rr * NC + t + n_lo] = acc[v];
        }
    }
}

// ---------------- GAT attention pieces ----------------
__global__ void k_attn_al(const float* __restrict__ h, const float* __restrict__ asrc,
                          const float* __restrict__ adst, float* __restrict__ alsrc,
                          float* __restrict__ aldst, int NC, int C) {
    int tid = blockIdx.x * blockDim.x + threadIdx.x;
    if (tid >= NN * 3) return;
    int nidx = tid / 3, hh = tid % 3;
    const float* hp = h + (size_t)nidx * NC + hh * C;
    float s = 0.0f, d = 0.0f;
    for (int c = 0; c < C; c++) { float v = hp[c]; s += v * asrc[hh * C + c]; d += v * adst[hh * C + c]; }
    alsrc[tid] = s; aldst[tid] = d;
}

__global__ void k_gat_max(const int* __restrict__ src, const int* __restrict__ dst,
                          const float* __restrict__ alsrc, const float* __restrict__ aldst,
                          float* __restrict__ ebuf, unsigned* __restrict__ mmax) {
    int idx = blockIdx.x * blockDim.x + threadIdx.x;
    if (idx >= EP * 3) return;
    int e = idx / 3, hh = idx % 3;
    int s, d;
    if (e < NE) { s = src[e]; d = dst[e]; } else { s = d = e - NE; }
    float v = lrelu02_(alsrc[s * 3 + hh] + aldst[d * 3 + hh]);
    ebuf[idx] = v;
    atomicMax(&mmax[d * 3 + hh], encf_(v));
}

__global__ void k_gat_exp(const int* __restrict__ dst, float* __restrict__ ebuf,
                          const unsigned* __restrict__ mmax, float* __restrict__ ssum) {
    int idx = blockIdx.x * blockDim.x + threadIdx.x;
    if (idx >= EP * 3) return;
    int e = idx / 3, hh = idx % 3;
    int d = (e < NE) ? dst[e] : (e - NE);
    float pv = __expf(ebuf[idx] - decf_(mmax[d * 3 + hh]));
    ebuf[idx] = pv;
    atomicAdd(&ssum[d * 3 + hh], pv);
}

__global__ void k_gat_acc(const int* __restrict__ src, const int* __restrict__ dst,
                          const float* __restrict__ ebuf, const float* __restrict__ ssum,
                          const float* __restrict__ h, float* __restrict__ out, int NC, int C) {
    int idx = blockIdx.x * blockDim.x + threadIdx.x;
    if (idx >= EP * 3) return;
    int e = idx / 3, hh = idx % 3;
    int s, d;
    if (e < NE) { s = src[e]; d = dst[e]; } else { s = d = e - NE; }
    float alpha = ebuf[idx] / (ssum[d * 3 + hh] + 1e-16f);
    const float* hp = h + (size_t)s * NC + hh * C;
    float* op = out + (size_t)d * NC + hh * C;
    for (int c = 0; c < C; c++) atomicAdd(&op[c], hp[c] * alpha);
}

__global__ void k_bias_relu(float* __restrict__ out, const float* __restrict__ b, int NC, int n) {
    int i = blockIdx.x * blockDim.x + threadIdx.x;
    if (i < n) { float v = out[i] + b[i % NC]; out[i] = v > 0.0f ? v : 0.0f; }
}

// ---------------- pooling + head ----------------
__global__ void k_pool(const float* __restrict__ go, const int* __restrict__ batch,
                       float* __restrict__ gsum, float* __restrict__ cnt) {
    int idx = blockIdx.x * blockDim.x + threadIdx.x;
    if (idx >= NN * 48) return;
    int nidx = idx / 48, c = idx % 48;
    int b = batch[nidx];
    atomicAdd(&gsum[b * 48 + c], go[idx]);
    if (c == 0) atomicAdd(&cnt[b], 1.0f);
}

__global__ void k_head(const float* __restrict__ gsum, const float* __restrict__ cnt,
                       const float* __restrict__ Wl1, const float* __restrict__ bl1,
                       const float* __restrict__ Wl2, const float* __restrict__ bl2,
                       float* __restrict__ out) {
    int g = threadIdx.x;
    if (g >= 64) return;
    float inv = 1.0f / fmaxf(cnt[g], 1.0f);
    float hid[16];
    for (int j = 0; j < 16; j++) {
        float s = bl1[j];
        for (int i = 0; i < 48; i++) s += gsum[g * 48 + i] * inv * Wl1[i * 16 + j];
        hid[j] = s > 0.0f ? s : 0.0f;
    }
    for (int k = 0; k < 10; k++) {
        float s = bl2[k];
        for (int j = 0; j < 16; j++) s += hid[j] * Wl2[j * 10 + k];
        out[g * 10 + k] = s;
    }
}

// ---------------- driver ----------------
extern "C" void kernel_launch(void* const* d_in, const int* in_sizes, int n_in,
                              void* d_out, int out_size, void* d_ws, size_t ws_size,
                              hipStream_t stream) {
    (void)in_sizes; (void)n_in; (void)out_size; (void)ws_size;
    const float* x    = (const float*)d_in[0];
    const float* ea   = (const float*)d_in[1];
    const int*   ei   = (const int*)d_in[2];
    const int*   srcI = ei;
    const int*   dstI = ei + NE;
    const int*   batch= (const int*)d_in[3];
    const float *Wf1=(const float*)d_in[4],  *bf1=(const float*)d_in[5];
    const float *Ws1=(const float*)d_in[6],  *bs1=(const float*)d_in[7];
    const float *Wf2=(const float*)d_in[8],  *bf2=(const float*)d_in[9];
    const float *Ws2=(const float*)d_in[10], *bs2=(const float*)d_in[11];
    const float *Wg1=(const float*)d_in[12], *as1=(const float*)d_in[13];
    const float *ad1=(const float*)d_in[14], *bg1=(const float*)d_in[15];
    const float *Wg2=(const float*)d_in[16], *as2=(const float*)d_in[17];
    const float *ad2=(const float*)d_in[18], *bg2=(const float*)d_in[19];
    const float *Wl1=(const float*)d_in[20], *bl1=(const float*)d_in[21];
    const float *Wl2=(const float*)d_in[22], *bl2=(const float*)d_in[23];

    float* ws = (float*)d_ws;
    size_t o = 0;
    float* agg  = ws + o; o += (size_t)32 * NN;
    float* h1   = ws + o; o += (size_t)32 * NN;
    float* h2   = ws + o; o += (size_t)32 * NN;
    float* gh   = ws + o; o += (size_t)96 * NN;
    float* go1  = ws + o; o += (size_t)96 * NN;
    float* ebuf = ws + o; o += (size_t)3 * EP;
    float* alsrc= ws + o; o += (size_t)3 * NN;
    float* aldst= ws + o; o += (size_t)3 * NN;
    unsigned* mmax = (unsigned*)(ws + o); o += (size_t)3 * NN;
    float* ssum = ws + o; o += (size_t)3 * NN;
    float* gsum = ws + o; o += 64 * 48;
    float* cnt  = ws + o; o += 64;
    _Float16* w16 = (_Float16*)(ws + o);
    _Float16* wf1h = w16;
    _Float16* ws1h = wf1h + 96 * 32;
    _Float16* wf2h = ws1h + 96 * 32;
    _Float16* ws2h = wf2h + 96 * 32;
    _Float16* wg1h = ws2h + 96 * 32;   // 32x96 col-major (Kpad=32)
    _Float16* wg2h = wg1h + 32 * 96;   // 96x48 col-major (Kpad=96)
    // region reuse after cgconv/gat1 retire:
    float* gh2 = gh;    // gat2 projected features, 48*NN <= 96*NN
    float* go2 = agg;   // gat2 output, 48*NN <= 64*NN (agg+h1, both free by then)

    const int T = 256;
    // weights -> f16 col-major
    k_convw<<<(96*32 + T-1)/T, T, 0, stream>>>(Wf1, wf1h, 72, 96, 32);
    k_convw<<<(96*32 + T-1)/T, T, 0, stream>>>(Ws1, ws1h, 72, 96, 32);
    k_convw<<<(96*32 + T-1)/T, T, 0, stream>>>(Wf2, wf2h, 72, 96, 32);
    k_convw<<<(96*32 + T-1)/T, T, 0, stream>>>(Ws2, ws2h, 72, 96, 32);
    k_convw<<<(32*96 + T-1)/T, T, 0, stream>>>(Wg1, wg1h, 32, 32, 96);
    k_convw<<<(96*48 + T-1)/T, T, 0, stream>>>(Wg2, wg2h, 96, 96, 48);

    const int edgeBlocks = (NE + 63) / 64;     // 16 edges/wave, 4 waves/block
    const int nodeBlocks = (NN + 63) / 64;

    // ---- CGConv1 ----
    hipMemsetAsync(agg, 0, (size_t)32 * NN * sizeof(float), stream);
    k_cgconv_edges<<<edgeBlocks, 128, 0, stream>>>(x, ea, srcI, dstI, wf1h, ws1h, bf1, bs1, agg);
    k_add_relu<<<(32*NN + T-1)/T, T, 0, stream>>>(agg, x, h1, 32 * NN);
    // ---- CGConv2 ----
    hipMemsetAsync(agg, 0, (size_t)32 * NN * sizeof(float), stream);
    k_cgconv_edges<<<edgeBlocks, 128, 0, stream>>>(h1, ea, srcI, dstI, wf2h, ws2h, bf2, bs2, agg);
    k_add_relu<<<(32*NN + T-1)/T, T, 0, stream>>>(agg, h1, h2, 32 * NN);

    const int eb3 = (EP * 3 + T - 1) / T;
    // ---- GAT1: 32 -> 3 heads x 32 (NC=96) ----
    k_gemm_wmma<<<nodeBlocks, 128, 0, stream>>>(h2, wg1h, gh, NN, 32, 96);
    k_attn_al<<<(NN*3 + T-1)/T, T, 0, stream>>>(gh, as1, ad1, alsrc, aldst, 96, 32);
    hipMemsetAsync(mmax, 0, (size_t)3 * NN * sizeof(unsigned), stream);
    hipMemsetAsync(ssum, 0, (size_t)3 * NN * sizeof(float), stream);
    hipMemsetAsync(go1, 0, (size_t)96 * NN * sizeof(float), stream);
    k_gat_max<<<eb3, T, 0, stream>>>(srcI, dstI, alsrc, aldst, ebuf, mmax);
    k_gat_exp<<<eb3, T, 0, stream>>>(dstI, ebuf, mmax, ssum);
    k_gat_acc<<<eb3, T, 0, stream>>>(srcI, dstI, ebuf, ssum, gh, go1, 96, 32);
    k_bias_relu<<<(96*NN + T-1)/T, T, 0, stream>>>(go1, bg1, 96, 96 * NN);

    // ---- GAT2: 96 -> 3 heads x 16 (NC=48) ----
    k_gemm_wmma<<<nodeBlocks, 128, 0, stream>>>(go1, wg2h, gh2, NN, 96, 48);
    k_attn_al<<<(NN*3 + T-1)/T, T, 0, stream>>>(gh2, as2, ad2, alsrc, aldst, 48, 16);
    hipMemsetAsync(mmax, 0, (size_t)3 * NN * sizeof(unsigned), stream);
    hipMemsetAsync(ssum, 0, (size_t)3 * NN * sizeof(float), stream);
    hipMemsetAsync(go2, 0, (size_t)48 * NN * sizeof(float), stream);
    k_gat_max<<<eb3, T, 0, stream>>>(srcI, dstI, alsrc, aldst, ebuf, mmax);
    k_gat_exp<<<eb3, T, 0, stream>>>(dstI, ebuf, mmax, ssum);
    k_gat_acc<<<eb3, T, 0, stream>>>(srcI, dstI, ebuf, ssum, gh2, go2, 48, 16);
    k_bias_relu<<<(48*NN + T-1)/T, T, 0, stream>>>(go2, bg2, 48, 48 * NN);

    // ---- pool + head ----
    hipMemsetAsync(gsum, 0, (64 * 48 + 64) * sizeof(float), stream);  // gsum + cnt contiguous
    k_pool<<<(NN*48 + T-1)/T, T, 0, stream>>>(go2, batch, gsum, cnt);
    k_head<<<1, 64, 0, stream>>>(gsum, cnt, Wl1, bl1, Wl2, bl2, (float*)d_out);
}
